// TorchRoughSABR_FMM_46961172415029
// MI455X (gfx1250) — matrix operand order repaired and checked by reading, not compiled, not run on hardware
//
#include <hip/hip_runtime.h>
#include <math.h>

// ---------------------------------------------------------------------------
// Rough-SABR FMM simulation for MI455X (gfx1250, wave32, WMMA).
//
// Stage 1: ndtri(u) -> dW_v (bf16), dW_r (f32)
// Stage 2: fractional kernel matrix -> pre-swizzled WMMA B fragments (bf16)
// Stage 3: fBm = dW_v @ K^T via v_wmma_f32_16x16x32_bf16, fused exp epilogue
//          -> V2_dt, dM stored [step][path] for coalesced scan loads
// Stage 4: per-path 128-step scan over 40 tenors (registers + LDS broadcast)
// ---------------------------------------------------------------------------

typedef __attribute__((ext_vector_type(16))) __bf16 v16bf;
typedef __attribute__((ext_vector_type(8)))  __bf16 v8bf;
typedef __attribute__((ext_vector_type(8)))  float  v8f;

#define NSTEPS 128      // reference N_STEPS
#define NTEN   40       // reference N (tenors)
#define HCONST 0.1f
#define GAMMA6 1.4891922488128171f   // Gamma(H + 0.5) = Gamma(0.6)

__device__ __forceinline__ unsigned short f2bf(float f) {
  // round-to-nearest-even float -> bf16
  unsigned int u = __float_as_uint(f);
  unsigned int r = u + 0x7FFFu + ((u >> 16) & 1u);
  return (unsigned short)(r >> 16);
}

// Acklam's inverse normal CDF approximation (single precision).
__device__ __forceinline__ float ndtri_f(float p) {
  const float a1 = -3.969683028665376e+01f, a2 =  2.209460984245205e+02f,
              a3 = -2.759285104469687e+02f, a4 =  1.383577518672690e+02f,
              a5 = -3.066479806614716e+01f, a6 =  2.506628277459239e+00f;
  const float b1 = -5.447609879822406e+01f, b2 =  1.615858368580409e+02f,
              b3 = -1.556989798598866e+02f, b4 =  6.680131188771972e+01f,
              b5 = -1.328068155288572e+01f;
  const float c1 = -7.784894002430293e-03f, c2 = -3.223964580411365e-01f,
              c3 = -2.400758277161838e+00f, c4 = -2.549732539343734e+00f,
              c5 =  4.374664141464968e+00f, c6 =  2.938163982698783e+00f;
  const float d1 =  7.784695709041462e-03f, d2 =  3.224671290700398e-01f,
              d3 =  2.445134137142996e+00f, d4 =  3.754408661907416e+00f;
  const float plow = 0.02425f, phigh = 1.0f - 0.02425f;
  if (p < plow) {
    float q = sqrtf(-2.0f * logf(p));
    return (((((c1*q+c2)*q+c3)*q+c4)*q+c5)*q+c6) /
           ((((d1*q+d2)*q+d3)*q+d4)*q+1.0f);
  } else if (p <= phigh) {
    float q = p - 0.5f, r = q * q;
    return (((((a1*r+a2)*r+a3)*r+a4)*r+a5)*r+a6)*q /
           (((((b1*r+b2)*r+b3)*r+b4)*r+b5)*r+1.0f);
  } else {
    float q = sqrtf(-2.0f * logf(1.0f - p));
    return -(((((c1*q+c2)*q+c3)*q+c4)*q+c5)*q+c6) /
            ((((d1*q+d2)*q+d3)*q+d4)*q+1.0f);
  }
}

// ---------------------------------------------------------------------------
// Stage 1: z = ndtri(clip(u)); dW_v = z0*sqdt (bf16); dW_r = (rho z0 + rbar z1)*sqdt
// ---------------------------------------------------------------------------
__global__ void prep_kernel(const float* __restrict__ u,
                            const float* __restrict__ tg,
                            const float* __restrict__ rhos,
                            unsigned short* __restrict__ dwv_bf,
                            float* __restrict__ dwr,
                            int total) {
  int i = blockIdx.x * blockDim.x + threadIdx.x;
  if (i >= total) return;
  float dt   = tg[1] - tg[0];
  float sqdt = sqrtf(dt);
  float rho  = rhos[0];
  float rbar = sqrtf(1.0f - rho * rho);
  float u0 = u[2 * i], u1 = u[2 * i + 1];
  u0 = fminf(fmaxf(u0, 1e-7f), 1.0f - 1e-7f);
  u1 = fminf(fmaxf(u1, 1e-7f), 1.0f - 1e-7f);
  float z0 = ndtri_f(u0);
  float z1 = ndtri_f(u1);
  dwv_bf[i] = f2bf(z0 * sqdt);
  dwr[i]    = (rho * z0 + rbar * z1) * sqdt;
}

// ---------------------------------------------------------------------------
// Stage 2: B[k][n] = kernel[n][k] = (tg[n+1]-tg[k])^(H-1/2)/Gamma for k<=n,
// stored pre-swizzled in WMMA 16-bit B-fragment layout (per ISA 7.12.2):
//   per (k-block kb, n-tile nt): lane = n % 16 + 16*((k>>3)&1) holds column n; v16bf element
//   e = (k&7) + 8*((k>>4)&1) (VGPR e/2, half e&1). One aligned 32B load per fragment.
// ---------------------------------------------------------------------------
__global__ void buildB_kernel(const float* __restrict__ tg,
                              unsigned short* __restrict__ bsw,
                              int NS) {
  int idx = blockIdx.x * blockDim.x + threadIdx.x;
  if (idx >= NS * NS) return;
  int k = idx / NS, n = idx % NS;
  float dtm = tg[n + 1] - tg[k];
  float val = 0.0f;
  if (dtm > 0.0f)
    val = powf(fmaxf(dtm, 1e-12f), HCONST - 0.5f) * (1.0f / GAMMA6);
  int kb = k >> 5, lane = (n & 15) + 16 * ((k >> 3) & 1), nt = n >> 4, e = (k & 7) + 8 * ((k >> 4) & 1);
  int NT = NS >> 4;
  bsw[((size_t)(kb * NT + nt) * 32 + lane) * 16 + e] = f2bf(val);
}

// ---------------------------------------------------------------------------
// Stage 3: one wave per 16x16 C tile. K swept in 4 chunks of 32 with
// v_wmma_f32_16x16x32_bf16. A fragment (16-bit A 16x32, ISA 7.12.2):
//   lane = {m = lane&15, half = lane>>4}; VGPRs 0..3 hold K = half*8 + 2v{,+1},
//   VGPRs 4..7 hold K = 16 + half*8 + 2v{,+1}  -> two contiguous 16B loads.
// Epilogue uses C layout: element r of v8f -> path m0 + r + 8*(lane>=16),
// step n0 + (lane&15).
// ---------------------------------------------------------------------------
__global__ void __launch_bounds__(32)
fbm_wmma_kernel(const unsigned short* __restrict__ dwv_bf,
                const unsigned short* __restrict__ bsw,
                const float* __restrict__ dwr,
                const float* __restrict__ tg,
                const float* __restrict__ nus,
                float* __restrict__ V2s,
                float* __restrict__ dMs,
                int P, int NS) {
  int lane = threadIdx.x;
  int m0 = blockIdx.x * 16, n0 = blockIdx.y * 16;
  int half = lane >> 4;
  int mrow = m0 + (lane & 15);
  int NT = NS >> 4, KB = NS >> 5;

  v8f acc = {};
  const unsigned short* arow = dwv_bf + (size_t)mrow * NS + half * 8;
  for (int kb = 0; kb < KB; ++kb) {
    v8bf alo = *(const v8bf*)(arow + kb * 32);
    v8bf ahi = *(const v8bf*)(arow + kb * 32 + 16);
    v16bf a = __builtin_shufflevector(alo, ahi,
        0, 1, 2, 3, 4, 5, 6, 7, 8, 9, 10, 11, 12, 13, 14, 15);
    v16bf b = *(const v16bf*)(bsw +
        ((size_t)(kb * NT + blockIdx.y) * 32 + lane) * 16);
    acc = __builtin_amdgcn_wmma_f32_16x16x32_bf16(
        false, a, false, b, (short)0, acc, false, false);
  }

  float dt = tg[1] - tg[0];
  float nu = nus[0];
  int step = n0 + (lane & 15);
  float t = tg[step + 1];
  float varc = 0.5f * nu * nu * powf(t, 2.0f * HCONST) /
               (2.0f * HCONST * GAMMA6 * GAMMA6);
#pragma unroll
  for (int r = 0; r < 8; ++r) {
    int p = m0 + r + 8 * half;
    float uv = expf(nu * acc[r] - varc);
    float v2dt = uv * uv * dt;
    float dm = uv * dwr[(size_t)p * NS + step];
    V2s[(size_t)step * P + p] = v2dt;   // [step][path] for coalesced scan
    dMs[(size_t)step * P + p] = dm;
  }
}

// ---------------------------------------------------------------------------
// Stage 4: per-path scan. drift_weights[j] = sum_{k>j} omega_k (old F), so a
// descending suffix-sum where omega_j is accumulated after being consumed.
// ---------------------------------------------------------------------------
__global__ void scan_kernel(const float* __restrict__ V2s,
                            const float* __restrict__ dMs,
                            const float* __restrict__ F0,
                            const float* __restrict__ tau,
                            const float* __restrict__ alphas,
                            float* __restrict__ out,
                            int P, int NS) {
  __shared__ float sal[NTEN], stau[NTEN], sta[NTEN], sF0[NTEN];
  for (int j = threadIdx.x; j < NTEN; j += blockDim.x) {
    sal[j]  = alphas[j];
    stau[j] = tau[j];
    sta[j]  = tau[j] * alphas[j];
    sF0[j]  = F0[j];
  }
  __syncthreads();
  int p = blockIdx.x * blockDim.x + threadIdx.x;
  if (p >= P) return;

  float F[NTEN];
  float* op = out + (size_t)p * (NS + 1) * NTEN;
#pragma unroll
  for (int j = 0; j < NTEN; ++j) { F[j] = sF0[j]; op[j] = F[j]; }

  for (int s = 0; s < NS; ++s) {
    float v2 = V2s[(size_t)s * P + p];
    float dm = dMs[(size_t)s * P + p];
    float* row = op + (size_t)(s + 1) * NTEN;
    float suffix = 0.0f;
#pragma unroll
    for (int j = NTEN - 1; j >= 0; --j) {
      float omega = sta[j] / (1.0f + stau[j] * F[j]);   // from old F
      float Fn = fmaf(sal[j], dm - suffix * v2, F[j]);
      row[j] = Fn;
      suffix += omega;
      F[j] = Fn;
    }
  }
}

// ---------------------------------------------------------------------------
extern "C" void kernel_launch(void* const* d_in, const int* in_sizes, int n_in,
                              void* d_out, int out_size, void* d_ws, size_t ws_size,
                              hipStream_t stream) {
  const float* u   = (const float*)d_in[0];   // (P, NS, 2)
  const float* tg  = (const float*)d_in[1];   // (NS+1,)
  const float* tau = (const float*)d_in[2];   // (N,)
  const float* F0  = (const float*)d_in[3];   // (N,)
  const float* al  = (const float*)d_in[4];   // (N,)
  const float* rho = (const float*)d_in[5];   // (N,)
  const float* nu  = (const float*)d_in[6];   // (N,)
  (void)n_in; (void)out_size; (void)ws_size;

  const int NS = in_sizes[1] - 1;             // 128
  const int P  = in_sizes[0] / (NS * 2);      // 16384

  // Workspace carve (all offsets 256B aligned; total ~29.4 MB)
  char* w = (char*)d_ws;
  size_t o = 0;
  unsigned short* dwv = (unsigned short*)(w + o); o += (size_t)P * NS * 2;   // bf16
  float*          dwr = (float*)(w + o);          o += (size_t)P * NS * 4;
  unsigned short* bsw = (unsigned short*)(w + o); o += (size_t)NS * NS * 2;  // bf16
  float*          V2s = (float*)(w + o);          o += (size_t)P * NS * 4;
  float*          dMs = (float*)(w + o);          o += (size_t)P * NS * 4;

  int total = P * NS;
  prep_kernel<<<(total + 255) / 256, 256, 0, stream>>>(u, tg, rho, dwv, dwr, total);
  buildB_kernel<<<(NS * NS + 255) / 256, 256, 0, stream>>>(tg, bsw, NS);
  fbm_wmma_kernel<<<dim3(P / 16, NS / 16), 32, 0, stream>>>(
      dwv, bsw, dwr, tg, nu, V2s, dMs, P, NS);
  scan_kernel<<<(P + 127) / 128, 128, 0, stream>>>(
      V2s, dMs, F0, tau, al, (float*)d_out, P, NS);
}
